// SharedMoEAudioProjector_18502719111703
// MI455X (gfx1250) — matrix operand order repaired and checked by reading, not compiled, not run on hardware
//
#include <hip/hip_runtime.h>
#include <math.h>
#include <stdint.h>

// ---------------------------------------------------------------------------
// SharedMoEAudioProjector for MI455X (gfx1250, wave32, WMMA bf16 16x16x32)
//
// B=16 S=2048 D=1280 K=4 -> N=8192 tokens, IN=5120, HID=1024, OUT=2048, E=4
// Compute-bound (~1.03 TFLOP/call) -> bf16 WMMA with f32 accumulation.
// GEMM K-tiles are streamed with GLOBAL_LOAD_ASYNC_TO_LDS_B128 (ASYNCcnt)
// into double-buffered LDS so the matrix pipe never waits on global latency.
// ---------------------------------------------------------------------------

#define TOK_N   8192
#define DIN     5120
#define DHID    1024
#define DOUT    2048
#define NEXP    4
#define EPSV    1e-6f

typedef __bf16 bf16;
typedef __attribute__((ext_vector_type(16))) __bf16 v16bf;
typedef __attribute__((ext_vector_type(8)))  __bf16 v8bf;
typedef __attribute__((ext_vector_type(8)))  float  v8f;

// LDS aperture rule (ISA 10.2): generic LDS address low 32 bits == LDS offset.
__device__ __forceinline__ uint32_t lds_addr32(const void* p) {
  return (uint32_t)(uintptr_t)p;
}
// CDNA5 async memory->LDS copy, 16B per lane, tracked by ASYNCcnt.
__device__ __forceinline__ void async_copy16(uint32_t ldsOff, const bf16* gptr) {
  asm volatile("global_load_async_to_lds_b128 %0, %1, off"
               :: "v"(ldsOff), "v"(gptr)
               : "memory");
}

// ---------------------------------------------------------------------------
// Weight convert + transpose:  W fp32 [K x N]  ->  Wt bf16 [N x K]
// out row = n*mult + off  (mult=2,off=0 gate / mult=2,off=1 up interleave)
// ---------------------------------------------------------------------------
__global__ __launch_bounds__(256) void transpose_cvt_k(
    const float* __restrict__ W, int Kd, int Nn,
    bf16* __restrict__ Wt, int mult, int off)
{
  __shared__ float tile[32][33];
  const int n0 = blockIdx.x * 32;
  const int k0 = blockIdx.y * 32;
  const int tx = threadIdx.x;        // 0..31
  const int ty = threadIdx.y;        // 0..7
#pragma unroll
  for (int i = 0; i < 4; ++i) {
    int k = k0 + ty + i * 8;
    tile[ty + i * 8][tx] = W[(size_t)k * Nn + n0 + tx];
  }
  __syncthreads();
#pragma unroll
  for (int i = 0; i < 4; ++i) {
    int n = n0 + ty + i * 8;
    size_t outr = (size_t)n * mult + off;
    Wt[outr * Kd + k0 + tx] = (bf16)tile[tx][ty + i * 8];
  }
}

// ---------------------------------------------------------------------------
// Frame-stack (pure reshape) + RMSNorm(ln_pre) + fp32->bf16
// ---------------------------------------------------------------------------
__global__ __launch_bounds__(256) void rms_pre_k(
    const float* __restrict__ x, const float* __restrict__ w,
    bf16* __restrict__ h)
{
  const int t = blockIdx.x, tid = threadIdx.x;
  const float* xr = x + (size_t)t * DIN;
  float v[DIN / 256];
  float ss = 0.f;
#pragma unroll
  for (int i = 0; i < DIN / 256; ++i) {
    v[i] = xr[tid + i * 256];
    ss += v[i] * v[i];
  }
#pragma unroll
  for (int o = 16; o > 0; o >>= 1) ss += __shfl_xor(ss, o, 32);
  __shared__ float red[8];
  if ((tid & 31) == 0) red[tid >> 5] = ss;
  __syncthreads();
  float tot = 0.f;
#pragma unroll
  for (int wv = 0; wv < 8; ++wv) tot += red[wv];
  const float rs = rsqrtf(tot / (float)DIN + EPSV);
  bf16* hr = h + (size_t)t * DIN;
#pragma unroll
  for (int i = 0; i < DIN / 256; ++i) {
    int idx = tid + i * 256;
    hr[idx] = (bf16)(v[i] * rs * w[idx]);
  }
}

// ---------------------------------------------------------------------------
// Router: logits = h @ router_w^T, softmax, top-2 renormalized -> combine[N,4]
// ---------------------------------------------------------------------------
__global__ __launch_bounds__(256) void router_k(
    const bf16* __restrict__ h, const float* __restrict__ rw,
    float* __restrict__ comb)
{
  const int t = blockIdx.x, tid = threadIdx.x;
  const bf16* hr = h + (size_t)t * DIN;
  float a0 = 0.f, a1 = 0.f, a2 = 0.f, a3 = 0.f;
  for (int i = tid; i < DIN; i += 256) {
    float hv = (float)hr[i];
    a0 += hv * rw[0 * DIN + i];
    a1 += hv * rw[1 * DIN + i];
    a2 += hv * rw[2 * DIN + i];
    a3 += hv * rw[3 * DIN + i];
  }
#pragma unroll
  for (int o = 16; o > 0; o >>= 1) {
    a0 += __shfl_xor(a0, o, 32); a1 += __shfl_xor(a1, o, 32);
    a2 += __shfl_xor(a2, o, 32); a3 += __shfl_xor(a3, o, 32);
  }
  __shared__ float red[8][4];
  if ((tid & 31) == 0) {
    int wv = tid >> 5;
    red[wv][0] = a0; red[wv][1] = a1; red[wv][2] = a2; red[wv][3] = a3;
  }
  __syncthreads();
  if (tid == 0) {
    float l[4] = {0.f, 0.f, 0.f, 0.f};
    for (int wv = 0; wv < 8; ++wv)
      for (int e = 0; e < 4; ++e) l[e] += red[wv][e];
    float mx = fmaxf(fmaxf(l[0], l[1]), fmaxf(l[2], l[3]));
    float p[4]; float s = 0.f;
    for (int e = 0; e < 4; ++e) { p[e] = __expf(l[e] - mx); s += p[e]; }
    for (int e = 0; e < 4; ++e) p[e] /= s;
    int i1 = 0;
    for (int e = 1; e < 4; ++e) if (p[e] > p[i1]) i1 = e;
    int i2 = -1;
    for (int e = 0; e < 4; ++e)
      if (e != i1 && (i2 < 0 || p[e] > p[i2])) i2 = e;
    float z = p[i1] + p[i2];
    float o[4] = {0.f, 0.f, 0.f, 0.f};
    o[i1] = p[i1] / z; o[i2] = p[i2] / z;
    float4 res = make_float4(o[0], o[1], o[2], o[3]);
    *(float4*)(comb + (size_t)t * 4) = res;
  }
}

// ---------------------------------------------------------------------------
// WMMA GEMM: C = A[M x Kd] (bf16, lda) @ Bt[Nout x Kd]^T (bf16, ldb)
// Block tile 128x128x64, 8 wave32 waves (2x4), wave tile 64x32 (4x2 WMMAs)
// Double-buffered LDS; K tiles streamed via async global->LDS (ASYNCcnt).
// mode 0: interleaved gate/up columns -> hid = silu(g)*u, bf16 [M x Nout/2]
// mode 1: Cacc[M x Nout] (+)= scale[row]*C  (scale==nullptr -> 1.0)
// ---------------------------------------------------------------------------
#define BM 128
#define BN 128
#define BK 64
#define LDSS (BK + 8)   // bf16 units, +16B pad per row

__device__ __forceinline__ v16bf frag_a(const bf16* base, int r16, int half, int kb) {
  // 16-bit A 16x32 layout: lanes 0-15 K={0..7,16..23}, lanes 16-31 K={8..15,24..31}
  union { v16bf v; v8bf h[2]; } u;
  const bf16* p = base + r16 * LDSS + kb;
  u.h[0] = *(const v8bf*)(p + half * 8);
  u.h[1] = *(const v8bf*)(p + 16 + half * 8);
  return u.v;
}
__device__ __forceinline__ v16bf frag_b(const bf16* base, int r16, int half, int kb) {
  // 16-bit B 32x16 layout: lanes 0-15 K={0..7,16..23}, lanes 16-31 K={8..15,24..31}, N=lane%16
  union { v16bf v; v8bf h[2]; } u;
  const bf16* p = base + r16 * LDSS + kb;
  u.h[0] = *(const v8bf*)(p + half * 8);
  u.h[1] = *(const v8bf*)(p + 16 + half * 8);
  return u.v;
}

__global__ __launch_bounds__(256, 1) void gemm_bf16_k(
    const bf16* __restrict__ A, int lda,
    const bf16* __restrict__ Bt, int ldb,
    int Kd, int mode,
    bf16* __restrict__ Chid, float* __restrict__ Cacc, int ldc,
    const float* __restrict__ scale, int scale_stride, int accumulate)
{
  __shared__ __align__(16) bf16 sA[2][BM * LDSS];
  __shared__ __align__(16) bf16 sB[2][BN * LDSS];

  const int tid  = threadIdx.x;
  const int lane = tid & 31, wave = tid >> 5;
  const int wm = wave >> 2, wn = wave & 3;      // wave grid 2(M) x 4(N)
  const int half = lane >> 4, r16 = lane & 15;
  const size_t m0 = (size_t)blockIdx.y * BM;
  const size_t n0 = (size_t)blockIdx.x * BN;

  // per-thread async-copy slots: 4 x 16B chunks for A and for B per K-tile
  uint32_t laOff[4], lbOff[4];
  const bf16* ga[4]; const bf16* gb[4];
#pragma unroll
  for (int i = 0; i < 4; ++i) {
    int c = tid + i * 256;
    int r = c >> 3, kc = c & 7;
    laOff[i] = lds_addr32(&sA[0][r * LDSS + kc * 8]);
    lbOff[i] = lds_addr32(&sB[0][r * LDSS + kc * 8]);
    ga[i] = A  + (m0 + r) * lda + kc * 8;
    gb[i] = Bt + (n0 + r) * ldb + kc * 8;
  }
  constexpr uint32_t BUFB = (uint32_t)(BM * LDSS * sizeof(bf16));

  v8f acc[4][2];
#pragma unroll
  for (int i = 0; i < 4; ++i)
#pragma unroll
    for (int j = 0; j < 2; ++j) acc[i][j] = (v8f)(0.f);

  const int nt = Kd / BK;
  // prologue: stream tile 0 into buffer 0 (8 async-copy instructions / wave)
#pragma unroll
  for (int i = 0; i < 4; ++i) {
    async_copy16(laOff[i], ga[i]);
    async_copy16(lbOff[i], gb[i]);
    ga[i] += BK; gb[i] += BK;
  }

  for (int t = 0; t < nt; ++t) {
    const int cur = t & 1;
    if (t + 1 < nt) {
      // stream next K-tile into the other buffer, then wait for the current
      // tile: async loads retire in order, so ASYNCcnt<=8 => tile t arrived.
      const uint32_t bofs = (cur ^ 1) ? BUFB : 0u;
#pragma unroll
      for (int i = 0; i < 4; ++i) {
        async_copy16(laOff[i] + bofs, ga[i]);
        async_copy16(lbOff[i] + bofs, gb[i]);
        ga[i] += BK; gb[i] += BK;
      }
      asm volatile("s_wait_asynccnt 0x8" ::: "memory");
    } else {
      asm volatile("s_wait_asynccnt 0x0" ::: "memory");
    }
    __syncthreads();

    const bf16* curA = sA[cur];
    const bf16* curB = sB[cur];
#pragma unroll
    for (int kk = 0; kk < 2; ++kk) {
      v16bf af[4], bfr[2];
#pragma unroll
      for (int i = 0; i < 4; ++i)
        af[i] = frag_a(curA + (wm * 64 + i * 16) * LDSS, r16, half, kk * 32);
#pragma unroll
      for (int j = 0; j < 2; ++j)
        bfr[j] = frag_b(curB + (wn * 32 + j * 16) * LDSS, r16, half, kk * 32);
#pragma unroll
      for (int i = 0; i < 4; ++i)
#pragma unroll
        for (int j = 0; j < 2; ++j)
          acc[i][j] = __builtin_amdgcn_wmma_f32_16x16x32_bf16(
              false, af[i], false, bfr[j], (short)0, acc[i][j], false, false);
    }
    __syncthreads();   // all waves done reading before buffer is re-streamed
  }

  if (mode == 1) {
    // out (+)= scale[row] * val
#pragma unroll
    for (int i = 0; i < 4; ++i) {
      size_t gr0 = m0 + wm * 64 + i * 16 + half * 8;
#pragma unroll
      for (int v = 0; v < 8; ++v) {
        size_t gr = gr0 + v;
        float s = scale ? scale[gr * scale_stride] : 1.0f;
#pragma unroll
        for (int j = 0; j < 2; ++j) {
          size_t gc = n0 + wn * 32 + j * 16 + r16;
          float val = acc[i][j][v] * s;
          float* dst = Cacc + gr * ldc + gc;
          if (accumulate) val += *dst;
          *dst = val;
        }
      }
    }
  } else {
    // fused SwiGLU: even columns = gate, odd = up (weights interleaved)
    const int parity = r16 & 1;
#pragma unroll
    for (int i = 0; i < 4; ++i) {
#pragma unroll
      for (int j = 0; j < 2; ++j) {
#pragma unroll
        for (int v = 0; v < 8; ++v) {
          float own = acc[i][j][v];
          float oth = __shfl_xor(own, 1, 32);   // partner column in lane^1
          if (parity == 0) {
            float g = own, u = oth;
            float hv = (g / (1.0f + __expf(-g))) * u;   // silu(g)*u
            size_t gr = m0 + wm * 64 + i * 16 + half * 8 + v;
            size_t gc = (n0 + wn * 32 + j * 16 + r16) >> 1;
            Chid[gr * ldc + gc] = (bf16)hv;
          }
        }
      }
    }
  }
}

// ---------------------------------------------------------------------------
// Final: out = rmsnorm((acc * layer_scale), ln_post_w)
// ---------------------------------------------------------------------------
__global__ __launch_bounds__(256) void rms_post_k(
    const float* __restrict__ accb, const float* __restrict__ ls,
    const float* __restrict__ w, float* __restrict__ out)
{
  const int t = blockIdx.x, tid = threadIdx.x;
  const float* ar = accb + (size_t)t * DOUT;
  float v[DOUT / 256];
  float ss = 0.f;
#pragma unroll
  for (int i = 0; i < DOUT / 256; ++i) {
    int idx = tid + i * 256;
    v[i] = ar[idx] * ls[idx];
    ss += v[i] * v[i];
  }
#pragma unroll
  for (int o = 16; o > 0; o >>= 1) ss += __shfl_xor(ss, o, 32);
  __shared__ float red[8];
  if ((tid & 31) == 0) red[tid >> 5] = ss;
  __syncthreads();
  float tot = 0.f;
#pragma unroll
  for (int wv = 0; wv < 8; ++wv) tot += red[wv];
  const float rs = rsqrtf(tot / (float)DOUT + EPSV);
  float* orow = out + (size_t)t * DOUT;
#pragma unroll
  for (int i = 0; i < DOUT / 256; ++i) {
    int idx = tid + i * 256;
    orow[idx] = v[i] * rs * w[idx];
  }
}

// ---------------------------------------------------------------------------
// Host launcher
// ---------------------------------------------------------------------------
extern "C" void kernel_launch(void* const* d_in, const int* in_sizes, int n_in,
                              void* d_out, int out_size, void* d_ws, size_t ws_size,
                              hipStream_t stream)
{
  (void)in_sizes; (void)n_in; (void)out_size; (void)ws_size;
  const float* x           = (const float*)d_in[0];
  const float* ln_pre_w    = (const float*)d_in[1];
  const float* ln_post_w   = (const float*)d_in[2];
  const float* router_w    = (const float*)d_in[3];
  const float* shared_gate = (const float*)d_in[4];
  const float* shared_up   = (const float*)d_in[5];
  const float* shared_down = (const float*)d_in[6];
  const float* exp_gate    = (const float*)d_in[7];
  const float* exp_up      = (const float*)d_in[8];
  const float* exp_down    = (const float*)d_in[9];
  const float* layer_scale = (const float*)d_in[10];
  float* outp = (float*)d_out;

  // workspace carve (~194 MB)
  char* p = (char*)d_ws;
  auto carve = [&](size_t bytes) -> char* {
    char* r = p; p += (bytes + 255) & ~(size_t)255; return r;
  };
  bf16*  h    = (bf16*)carve((size_t)TOK_N * DIN * sizeof(bf16));          // 84 MB
  bf16*  wgu  = (bf16*)carve((size_t)2 * DHID * DIN * sizeof(bf16));       // 21 MB
  bf16*  wd   = (bf16*)carve((size_t)DOUT * DHID * sizeof(bf16));          //  4 MB
  bf16*  hid  = (bf16*)carve((size_t)TOK_N * DHID * sizeof(bf16));         // 17 MB
  float* accb = (float*)carve((size_t)TOK_N * DOUT * sizeof(float));       // 67 MB
  float* comb = (float*)carve((size_t)TOK_N * NEXP * sizeof(float));       // .13 MB

  // 1) stacked RMSNorm -> bf16 activations
  rms_pre_k<<<TOK_N, 256, 0, stream>>>(x, ln_pre_w, h);
  // 2) router -> dense combine weights [N,4]
  router_k<<<TOK_N, 256, 0, stream>>>(h, router_w, comb);

  const dim3 tb(32, 8);
  const dim3 gemm_grid1(DOUT / BN, TOK_N / BM);   // Nout = 2*HID = 2048
  const dim3 gemm_grid2(DOUT / BN, TOK_N / BM);   // Nout = OUT  = 2048

  // 3) 5 SwiGLU units: shared first (writes acc), then 4 experts (accumulate)
  for (int u = 0; u < 5; ++u) {
    const float *g, *up, *dn, *sc;
    int accum;
    if (u == 0) {
      g = shared_gate; up = shared_up; dn = shared_down;
      sc = nullptr; accum = 0;
    } else {
      int e = u - 1;
      g  = exp_gate + (size_t)e * DIN * DHID;
      up = exp_up   + (size_t)e * DIN * DHID;
      dn = exp_down + (size_t)e * DHID * DOUT;
      sc = comb + e; accum = 1;
    }
    // convert+transpose (gate/up interleaved by output column)
    transpose_cvt_k<<<dim3(DHID / 32, DIN / 32), tb, 0, stream>>>(
        g,  DIN, DHID, wgu, 2, 0);
    transpose_cvt_k<<<dim3(DHID / 32, DIN / 32), tb, 0, stream>>>(
        up, DIN, DHID, wgu, 2, 1);
    transpose_cvt_k<<<dim3(DOUT / 32, DHID / 32), tb, 0, stream>>>(
        dn, DHID, DOUT, wd, 1, 0);
    // gate/up GEMM + fused SwiGLU -> hid bf16 [N, HID]
    gemm_bf16_k<<<gemm_grid1, 256, 0, stream>>>(
        h, DIN, wgu, DIN, DIN, /*mode=*/0,
        hid, nullptr, DHID, nullptr, 0, 0);
    // down GEMM, per-token routed scale, (+)= into fp32 accumulator
    gemm_bf16_k<<<gemm_grid2, 256, 0, stream>>>(
        hid, DHID, wd, DHID, DHID, /*mode=*/1,
        nullptr, accb, DOUT, sc, NEXP, accum);
  }

  // 4) layer_scale + RMSNorm(ln_post) -> fp32 output
  rms_post_k<<<TOK_N, 256, 0, stream>>>(accb, layer_scale, ln_post_w, outp);
}
